// Bottleneck_MHSA_48936857370685
// MI455X (gfx1250) — matrix operand order repaired and checked by reading, not compiled, not run on hardware
//
#include <hip/hip_runtime.h>
#include <hip/hip_bf16.h>
#include <math.h>

// ---------------------------------------------------------------------------
// Bottleneck-MHSA block for gfx1250 (MI455X), wave32 + WMMA bf16.
//   y1  = relu(bn1(W1 @ x))                      (256 x 1024 GEMM)
//   q,k,v = Wq/Wk/Wv @ y1 + b                    (256 x 256 GEMMs)
//   S = [q;rel]^T [k;q]  (K=128 augmented GEMM), flash softmax, O = P @ V^T
//   y3  = bn3(W3 @ relu(bn2(O)))                 (1024 x 256 GEMM)
//   out = relu(y3 + x)
// GEMMs: v_wmma_f32_16x16x32_bf16; operand staging via async global->LDS DMA
// (ASYNCcnt / s_wait_asynccnt) when the toolchain exposes the builtin.
// ---------------------------------------------------------------------------

typedef __bf16 bf16_t;
typedef bf16_t v16bf __attribute__((ext_vector_type(16)));
typedef float  v8f   __attribute__((ext_vector_type(8)));
typedef int    v4i   __attribute__((ext_vector_type(4)));

#define DEVINL __device__ __forceinline__

constexpr int   BATCH = 4;
constexpr int   CIN   = 1024;
constexpr int   CP    = 256;              // planes
constexpr int   NSP   = 14 * 14 * 14;     // 2744 tokens per batch
constexpr int   TTOK  = BATCH * NSP;      // 10976 total tokens (div by 32)
constexpr int   NP    = 2752;             // padded per-batch tokens (43*64)
constexpr int   DH    = 64;               // head dim
constexpr float EPSV  = 1e-5f;

union FragB { v16bf v; float4 q[2]; };
union AccF  { v8f  v; float  f[8]; };

#if __has_builtin(__builtin_amdgcn_global_load_async_to_lds_b128)
#define HAS_ASYNC_LDS 1
#else
#define HAS_ASYNC_LDS 0
#endif

// 16-byte global -> LDS copy; async DMA (ASYNCcnt) when available.
DEVINL void async_copy_b128(bf16_t* lds_dst, const bf16_t* gsrc) {
#if HAS_ASYNC_LDS
  __builtin_amdgcn_global_load_async_to_lds_b128(
      (__attribute__((address_space(1))) v4i*)(v4i*)(void*)gsrc,
      (__attribute__((address_space(3))) v4i*)(v4i*)(void*)lds_dst, 0, 0);
#else
  *(float4*)lds_dst = *(const float4*)gsrc;
#endif
}

DEVINL void wait_async() {
#if HAS_ASYNC_LDS
#if __has_builtin(__builtin_amdgcn_s_wait_asynccnt)
  __builtin_amdgcn_s_wait_asynccnt(0);
#else
  asm volatile("s_wait_asynccnt 0x0" ::: "memory");
#endif
#endif
}

// A fragment: 16x32 bf16 tile, row-major, leading dim ld.
// lane = row (lane&15); lanes 0-15 hold K {0..7,16..23}, lanes 16-31 {8..15,24..31}.
DEVINL v16bf load_a_frag(const bf16_t* tile, int ld, int lane) {
  const int m = lane & 15, g = lane >> 4;
  const bf16_t* p = tile + (size_t)m * ld + g * 8;
  FragB f;
  f.q[0] = *(const float4*)(p);        // K = g*8 .. g*8+7
  f.q[1] = *(const float4*)(p + 16);   // K = 16+g*8 ..
  return f.v;
}

// B fragment: 32x16 bf16 tile where column n is contiguous over K in memory
// (source stored "token-major" [n][k], leading dim ld).
DEVINL v16bf load_b_frag(const bf16_t* tile, int ld, int lane) {
  const int n = lane & 15, g = lane >> 4;
  const bf16_t* p = tile + (size_t)n * ld + g * 8;
  FragB f;
  f.q[0] = *(const float4*)(p);
  f.q[1] = *(const float4*)(p + 16);
  return f.v;
}

DEVINL v8f wmma_bf16(v16bf a, v16bf b, v8f c) {
  return __builtin_amdgcn_wmma_f32_16x16x32_bf16(
      /*neg_a=*/false, a, /*neg_b=*/false, b,
      /*c_mod=*/(short)0, c, /*reuse_a=*/false, /*reuse_b=*/false);
}

// ---------------------------------------------------------------------------
// Generic GEMM: C[T x M] (token-major fp32) = A[M x K] (bf16 row-major)
//               @ B (bf16 token-major [T x K]), optional per-M bias.
// Block: 256 thr = 8 waves stacked in M (128 rows), 32 tokens per block.
// B staged K=128 deep through LDS via async DMA; 8 WMMAs per barrier pair.
// ---------------------------------------------------------------------------
constexpr int KT = 128;

__global__ __launch_bounds__(256)
void gemm_wmma_kernel(const bf16_t* __restrict__ A, const bf16_t* __restrict__ Bt,
                      const float* __restrict__ bias, float* __restrict__ C,
                      int M, int K) {
  __shared__ alignas(32) bf16_t Bs[32 * KT];   // 8 KB [token-local][k-local]
  const int lane = threadIdx.x & 31;
  const int wave = threadIdx.x >> 5;
  const int m0 = (blockIdx.x * 8 + wave) * 16;
  const int n0 = blockIdx.y * 32;

  const int stok = threadIdx.x >> 3;           // 0..31 token row
  const int skk  = (threadIdx.x & 7) * 16;     // 0..112 k offset

  v8f acc0 = {}, acc1 = {};
  for (int k0 = 0; k0 < K; k0 += KT) {
    { // async stage: 32 tokens x 128 k = 8 KB, 32 B per thread
      const bf16_t* src = Bt + (size_t)(n0 + stok) * K + k0 + skk;
      bf16_t* dst = Bs + stok * KT + skk;
      async_copy_b128(dst, src);
      async_copy_b128(dst + 8, src + 8);
    }
    if (k0 + KT < K)  // warm caches for next A tile (global_prefetch_b8)
      __builtin_prefetch(A + (size_t)m0 * K + k0 + KT, 0, 1);
    wait_async();
    __syncthreads();
#pragma unroll
    for (int kk = 0; kk < KT; kk += 32) {
      v16bf a  = load_a_frag(A + (size_t)m0 * K + k0 + kk, K, lane);
      v16bf b0 = load_b_frag(Bs + kk, KT, lane);
      v16bf b1 = load_b_frag(Bs + 16 * KT + kk, KT, lane);
      acc0 = wmma_bf16(a, b0, acc0);
      acc1 = wmma_bf16(a, b1, acc1);
    }
    __syncthreads();
  }

  const int colt  = lane & 15;
  const int rbase = (lane >> 4) * 8;
  float bv[8];
#pragma unroll
  for (int r = 0; r < 8; ++r) bv[r] = bias ? bias[m0 + rbase + r] : 0.f;
  AccF A0, A1; A0.v = acc0; A1.v = acc1;
  float* d0 = C + (size_t)(n0 + colt) * M + m0 + rbase;
  float* d1 = C + (size_t)(n0 + 16 + colt) * M + m0 + rbase;
#pragma unroll
  for (int r = 0; r < 8; ++r) { d0[r] = A0.f[r] + bv[r]; d1[r] = A1.f[r] + bv[r]; }
}

// ---------------------------------------------------------------------------
// Fused flash attention per (b,h): S = augA @ augB^T (K=128), online softmax,
// O += P @ V^T.  One wave owns 16 query rows; 4 waves per block share the
// augB / V tiles through LDS (async-staged once per 32-key step).
// ---------------------------------------------------------------------------
__global__ __launch_bounds__(128)
void k_attention(const bf16_t* __restrict__ augA, const bf16_t* __restrict__ augB,
                 const bf16_t* __restrict__ vmat, float* __restrict__ att) {
  __shared__ alignas(32) bf16_t psh[4][16 * 32];   // per-wave P tiles (4 KB)
  __shared__ alignas(32) bf16_t BsB[32 * 128];     // augB tile (8 KB)
  __shared__ alignas(32) bf16_t BsV[64 * 32];      // V tile, d-major (4 KB)
  const int tid  = threadIdx.x;
  const int lane = tid & 31;
  const int wave = tid >> 5;
  const int bh = blockIdx.y;
  const int b = bh >> 2, h = bh & 3;
  const int n0 = blockIdx.x * 64 + wave * 16;

  const bf16_t* Ab = augA + ((size_t)bh * NP + n0) * 128;
  const bf16_t* Bb = augB + (size_t)bh * NP * 128;
  const bf16_t* Vb = vmat + (size_t)bh * DH * NP;
  bf16_t* ps = psh[wave];

  // staging coordinates (128 threads)
  const int btok = tid >> 2, bkk = (tid & 3) * 32;  // augB: 32 rows x 128, 64 B/thr
  const int vd   = tid >> 1, vmm = (tid & 1) * 16;  // V:    64 rows x 32, 32 B/thr

  // A fragments for this wave's 16 rows are loop-invariant (K = 128 -> 4 frags)
  v16bf afr[4];
#pragma unroll
  for (int kk = 0; kk < 4; ++kk) afr[kk] = load_a_frag(Ab + kk * 32, 128, lane);

  AccF O[4];
#pragma unroll
  for (int j = 0; j < 4; ++j) { v8f z = {}; O[j].v = z; }
  float mrow[8], lrow[8];
#pragma unroll
  for (int r = 0; r < 8; ++r) { mrow[r] = -1e30f; lrow[r] = 0.f; }

  const int colA  = lane & 15;
  const int rbase = (lane >> 4) * 8;

  for (int m0 = 0; m0 < NP; m0 += 32) {
    { // async stage augB rows m0..m0+31 and V columns m0..m0+31
      const bf16_t* sB = Bb + (size_t)(m0 + btok) * 128 + bkk;
      bf16_t* dB = BsB + btok * 128 + bkk;
      async_copy_b128(dB,      sB);
      async_copy_b128(dB + 8,  sB + 8);
      async_copy_b128(dB + 16, sB + 16);
      async_copy_b128(dB + 24, sB + 24);
      const bf16_t* sV = Vb + (size_t)vd * NP + m0 + vmm;
      bf16_t* dV = BsV + vd * 32 + vmm;
      async_copy_b128(dV,     sV);
      async_copy_b128(dV + 8, sV + 8);
    }
    wait_async();
    __syncthreads();

    // ---- scores for 32 key columns (two 16x16 tiles, shared A frags) ----
    v8f s0 = {}, s1 = {};
#pragma unroll
    for (int kk = 0; kk < 4; ++kk) {
      v16bf b0 = load_b_frag(BsB + kk * 32, 128, lane);
      v16bf b1 = load_b_frag(BsB + 16 * 128 + kk * 32, 128, lane);
      s0 = wmma_bf16(afr[kk], b0, s0);
      s1 = wmma_bf16(afr[kk], b1, s1);
    }
    AccF S0, S1; S0.v = s0; S1.v = s1;
    const bool v0 = (m0 + colA) < NSP;
    const bool v1 = (m0 + 16 + colA) < NSP;

    // ---- row max across the 16 lanes of each half-wave ----
    float tm[8];
#pragma unroll
    for (int r = 0; r < 8; ++r) {
      if (!v0) S0.f[r] = -1e30f;
      if (!v1) S1.f[r] = -1e30f;
      tm[r] = fmaxf(S0.f[r], S1.f[r]);
    }
#pragma unroll
    for (int st = 1; st < 16; st <<= 1)
#pragma unroll
      for (int r = 0; r < 8; ++r) tm[r] = fmaxf(tm[r], __shfl_xor(tm[r], st, 32));

    // ---- online softmax update ----
    float sc[8], rs[8];
#pragma unroll
    for (int r = 0; r < 8; ++r) {
      float nm = fmaxf(mrow[r], tm[r]);
      sc[r] = __expf(mrow[r] - nm);
      mrow[r] = nm;
      S0.f[r] = v0 ? __expf(S0.f[r] - nm) : 0.f;
      S1.f[r] = v1 ? __expf(S1.f[r] - nm) : 0.f;
      rs[r] = S0.f[r] + S1.f[r];
    }
#pragma unroll
    for (int st = 1; st < 16; st <<= 1)
#pragma unroll
      for (int r = 0; r < 8; ++r) rs[r] += __shfl_xor(rs[r], st, 32);

#pragma unroll
    for (int r = 0; r < 8; ++r) {
      lrow[r] = lrow[r] * sc[r] + rs[r];
      O[0].f[r] *= sc[r]; O[1].f[r] *= sc[r];
      O[2].f[r] *= sc[r]; O[3].f[r] *= sc[r];
      // stage P tile (C-layout -> 16x32 row-major bf16) for the A-operand
      ps[(rbase + r) * 32 + colA]      = (bf16_t)S0.f[r];
      ps[(rbase + r) * 32 + 16 + colA] = (bf16_t)S1.f[r];
    }
    asm volatile("s_wait_dscnt 0" ::: "memory");   // cross-lane LDS RAW in-wave

    // ---- O += P @ V^T : A = P (16n x 32m), B cols = head dims ----
    v16bf pfr = load_a_frag(ps, 32, lane);
#pragma unroll
    for (int j = 0; j < 4; ++j) {
      v16bf vb = load_b_frag(BsV + (j * 16) * 32, 32, lane);
      O[j].v = wmma_bf16(pfr, vb, O[j].v);
    }
    __syncthreads();   // protect BsB/BsV before next stage
  }

  // ---- normalize and store token-major [tok x 256] ----
#pragma unroll
  for (int r = 0; r < 8; ++r) {
    const int n = n0 + rbase + r;
    if (n < NSP) {
      const float inv = 1.f / lrow[r];
      float* dst = att + (size_t)(b * NSP + n) * CP + h * DH;
#pragma unroll
      for (int j = 0; j < 4; ++j) dst[j * 16 + colA] = O[j].f[r] * inv;
    }
  }
}

// ---------------------------------------------------------------------------
// Elementwise / reduction helper kernels
// ---------------------------------------------------------------------------
__global__ void k_x_tokmajor(const float* __restrict__ x, bf16_t* __restrict__ xb) {
  size_t i = (size_t)blockIdx.x * blockDim.x + threadIdx.x;
  if (i >= (size_t)BATCH * CIN * NSP) return;
  int s = (int)(i % NSP);
  int c = (int)((i / NSP) % CIN);
  int b = (int)(i / ((size_t)NSP * CIN));
  xb[(size_t)(b * NSP + s) * CIN + c] = (bf16_t)x[i];
}

__global__ void k_cvt_bf16(const float* __restrict__ src, bf16_t* __restrict__ dst, int n) {
  int i = blockIdx.x * blockDim.x + threadIdx.x;
  if (i < n) dst[i] = (bf16_t)src[i];
}

__global__ __launch_bounds__(256)
void k_bn_stats(const float* __restrict__ Y, int C,
                float* __restrict__ sum, float* __restrict__ sumsq) {
  const int c = blockIdx.x;
  float s = 0.f, s2 = 0.f;
  for (int t = threadIdx.x; t < TTOK; t += 256) {
    float v = Y[(size_t)t * C + c];
    s += v; s2 += v * v;
  }
  __shared__ float rs[256], rq[256];
  rs[threadIdx.x] = s; rq[threadIdx.x] = s2;
  __syncthreads();
  for (int off = 128; off > 0; off >>= 1) {
    if (threadIdx.x < off) {
      rs[threadIdx.x] += rs[threadIdx.x + off];
      rq[threadIdx.x] += rq[threadIdx.x + off];
    }
    __syncthreads();
  }
  if (threadIdx.x == 0) { sum[c] = rs[0]; sumsq[c] = rq[0]; }
}

__global__ void k_bn_relu_bf16(const float* __restrict__ Y, const float* __restrict__ gam,
                               const float* __restrict__ bet, const float* __restrict__ sum,
                               const float* __restrict__ sumsq, int C,
                               bf16_t* __restrict__ out) {
  size_t i = (size_t)blockIdx.x * blockDim.x + threadIdx.x;
  if (i >= (size_t)TTOK * C) return;
  int c = (int)(i % C);
  float mean = sum[c] * (1.f / (float)TTOK);
  float var  = sumsq[c] * (1.f / (float)TTOK) - mean * mean;
  float sc = gam[c] * rsqrtf(var + EPSV);
  float sh = bet[c] - mean * sc;
  out[i] = (bf16_t)fmaxf(Y[i] * sc + sh, 0.f);
}

// augA[n][0:64]=q[d,n], [64:128]=rel[h][d][n]; augB[m][0:64]=k, [64:128]=q
__global__ void k_pack_aug(const float* __restrict__ qf, const float* __restrict__ kf,
                           const float* __restrict__ rel_h, const float* __restrict__ rel_w,
                           const float* __restrict__ rel_d,
                           bf16_t* __restrict__ augA, bf16_t* __restrict__ augB) {
  size_t i = (size_t)blockIdx.x * blockDim.x + threadIdx.x;
  if (i >= (size_t)16 * NP * 128) return;
  int j  = (int)(i & 127);
  int n  = (int)((i >> 7) % NP);
  int bh = (int)(i / ((size_t)NP * 128));
  int b = bh >> 2, h = bh & 3;
  bf16_t va = (bf16_t)0.f, vb = (bf16_t)0.f;
  if (n < NSP) {
    int d = j & 63;
    size_t tc = (size_t)(b * NSP + n) * CP + h * DH + d;
    float q = qf[tc];
    if (j < 64) {
      va = (bf16_t)q;
      vb = (bf16_t)kf[tc];
    } else {
      int w = n / 196, hh = (n / 14) % 14, dd = n % 14;
      int base = h * DH + d;
      float r = rel_h[(base * 14 + hh) * 14 + dd]
              + rel_w[(base * 14 + w) * 14 + dd]
              + rel_d[(base * 14 + w) * 14 + hh];
      va = (bf16_t)r;
      vb = (bf16_t)q;
    }
  }
  augA[i] = va; augB[i] = vb;
}

// vmat[bh][d][m] = v[d, m]  (d-major so P@V B-fragments are contiguous in m)
__global__ void k_pack_v(const float* __restrict__ vf, bf16_t* __restrict__ vm) {
  size_t i = (size_t)blockIdx.x * blockDim.x + threadIdx.x;
  if (i >= (size_t)16 * DH * NP) return;
  int m  = (int)(i % NP);
  int d  = (int)((i / NP) % DH);
  int bh = (int)(i / ((size_t)NP * DH));
  int b = bh >> 2, h = bh & 3;
  vm[i] = (m < NSP) ? (bf16_t)vf[(size_t)(b * NSP + m) * CP + h * DH + d] : (bf16_t)0.f;
}

__global__ void k_final(const float* __restrict__ y3, const float* __restrict__ x,
                        const float* __restrict__ gam, const float* __restrict__ bet,
                        const float* __restrict__ sum, const float* __restrict__ sumsq,
                        float* __restrict__ out) {
  size_t i = (size_t)blockIdx.x * blockDim.x + threadIdx.x;
  if (i >= (size_t)BATCH * CIN * NSP) return;
  int s = (int)(i % NSP);
  int c = (int)((i / NSP) % CIN);
  int b = (int)(i / ((size_t)NSP * CIN));
  float mean = sum[c] * (1.f / (float)TTOK);
  float var  = sumsq[c] * (1.f / (float)TTOK) - mean * mean;
  float sc = gam[c] * rsqrtf(var + EPSV);
  float sh = bet[c] - mean * sc;
  float v = y3[(size_t)(b * NSP + s) * CIN + c] * sc + sh + x[i];
  out[i] = fmaxf(v, 0.f);
}

// ---------------------------------------------------------------------------
extern "C" void kernel_launch(void* const* d_in, const int* in_sizes, int n_in,
                              void* d_out, int out_size, void* d_ws, size_t ws_size,
                              hipStream_t stream) {
  const float* x     = (const float*)d_in[0];
  const float* W1    = (const float*)d_in[1];
  const float* g1    = (const float*)d_in[2];
  const float* b1    = (const float*)d_in[3];
  const float* Wq    = (const float*)d_in[4];
  const float* bq    = (const float*)d_in[5];
  const float* Wk    = (const float*)d_in[6];
  const float* bk    = (const float*)d_in[7];
  const float* Wv    = (const float*)d_in[8];
  const float* bv    = (const float*)d_in[9];
  const float* rel_h = (const float*)d_in[10];
  const float* rel_w = (const float*)d_in[11];
  const float* rel_d = (const float*)d_in[12];
  const float* g2    = (const float*)d_in[13];
  const float* b2    = (const float*)d_in[14];
  const float* W3    = (const float*)d_in[15];
  const float* g3    = (const float*)d_in[16];
  const float* b3    = (const float*)d_in[17];
  float* out = (float*)d_out;
  (void)in_sizes; (void)n_in; (void)out_size; (void)ws_size;

  char* ws = (char*)d_ws;
  size_t off = 0;
  auto alloc = [&](size_t bytes) -> char* {
    char* p = ws + off;
    off += (bytes + 255) & ~(size_t)255;
    return p;
  };

  bf16_t* xb   = (bf16_t*)alloc((size_t)TTOK * CIN * 2);
  bf16_t* w1b  = (bf16_t*)alloc((size_t)CP * CIN * 2);
  bf16_t* wqb  = (bf16_t*)alloc((size_t)CP * CP * 2);
  bf16_t* wkb  = (bf16_t*)alloc((size_t)CP * CP * 2);
  bf16_t* wvb  = (bf16_t*)alloc((size_t)CP * CP * 2);
  bf16_t* w3b  = (bf16_t*)alloc((size_t)CIN * CP * 2);
  float*  y1   = (float*)alloc((size_t)TTOK * CP * 4);
  bf16_t* y1b  = (bf16_t*)alloc((size_t)TTOK * CP * 2);
  float*  qf   = (float*)alloc((size_t)TTOK * CP * 4);
  float*  kf   = (float*)alloc((size_t)TTOK * CP * 4);
  float*  vf   = (float*)alloc((size_t)TTOK * CP * 4);
  bf16_t* augA = (bf16_t*)alloc((size_t)16 * NP * 128 * 2);
  bf16_t* augB = (bf16_t*)alloc((size_t)16 * NP * 128 * 2);
  bf16_t* vm   = (bf16_t*)alloc((size_t)16 * DH * NP * 2);
  float*  att  = (float*)alloc((size_t)TTOK * CP * 4);
  bf16_t* attb = (bf16_t*)alloc((size_t)TTOK * CP * 2);
  float*  y3   = (float*)alloc((size_t)TTOK * CIN * 4);
  float*  sum1 = (float*)alloc(CP * 4);
  float*  sq1  = (float*)alloc(CP * 4);
  float*  sum2 = (float*)alloc(CP * 4);
  float*  sq2  = (float*)alloc(CP * 4);
  float*  sum3 = (float*)alloc(CIN * 4);
  float*  sq3  = (float*)alloc(CIN * 4);

  const int EW = 256;
  const size_t nx = (size_t)BATCH * CIN * NSP;

  // --- input / weight conversion to bf16 (token-major for GEMM B operand) ---
  k_x_tokmajor<<<(unsigned)((nx + EW - 1) / EW), EW, 0, stream>>>(x, xb);
  k_cvt_bf16<<<(CP * CIN + EW - 1) / EW, EW, 0, stream>>>(W1, w1b, CP * CIN);
  k_cvt_bf16<<<(CP * CP + EW - 1) / EW, EW, 0, stream>>>(Wq, wqb, CP * CP);
  k_cvt_bf16<<<(CP * CP + EW - 1) / EW, EW, 0, stream>>>(Wk, wkb, CP * CP);
  k_cvt_bf16<<<(CP * CP + EW - 1) / EW, EW, 0, stream>>>(Wv, wvb, CP * CP);
  k_cvt_bf16<<<(CIN * CP + EW - 1) / EW, EW, 0, stream>>>(W3, w3b, CIN * CP);

  // --- conv1 + BN1 + ReLU ---
  dim3 gc1(CP / 128, TTOK / 32);
  gemm_wmma_kernel<<<gc1, 256, 0, stream>>>(w1b, xb, nullptr, y1, CP, CIN);
  k_bn_stats<<<CP, 256, 0, stream>>>(y1, CP, sum1, sq1);
  k_bn_relu_bf16<<<(unsigned)(((size_t)TTOK * CP + EW - 1) / EW), EW, 0, stream>>>(
      y1, g1, b1, sum1, sq1, CP, y1b);

  // --- QKV projections ---
  dim3 gqkv(CP / 128, TTOK / 32);
  gemm_wmma_kernel<<<gqkv, 256, 0, stream>>>(wqb, y1b, bq, qf, CP, CP);
  gemm_wmma_kernel<<<gqkv, 256, 0, stream>>>(wkb, y1b, bk, kf, CP, CP);
  gemm_wmma_kernel<<<gqkv, 256, 0, stream>>>(wvb, y1b, bv, vf, CP, CP);

  // --- pack augmented operands + V, run fused flash attention ---
  const size_t naug = (size_t)16 * NP * 128;
  k_pack_aug<<<(unsigned)((naug + EW - 1) / EW), EW, 0, stream>>>(
      qf, kf, rel_h, rel_w, rel_d, augA, augB);
  const size_t nvm = (size_t)16 * DH * NP;
  k_pack_v<<<(unsigned)((nvm + EW - 1) / EW), EW, 0, stream>>>(vf, vm);
  dim3 ga(NP / 64, 16);
  k_attention<<<ga, 128, 0, stream>>>(augA, augB, vm, att);

  // --- BN2 + ReLU, W3 GEMM, BN3 + residual + ReLU ---
  k_bn_stats<<<CP, 256, 0, stream>>>(att, CP, sum2, sq2);
  k_bn_relu_bf16<<<(unsigned)(((size_t)TTOK * CP + EW - 1) / EW), EW, 0, stream>>>(
      att, g2, b2, sum2, sq2, CP, attb);
  dim3 gc3(CIN / 128, TTOK / 32);
  gemm_wmma_kernel<<<gc3, 256, 0, stream>>>(w3b, attb, nullptr, y3, CIN, CP);
  k_bn_stats<<<CIN, 256, 0, stream>>>(y3, CIN, sum3, sq3);
  k_final<<<(unsigned)((nx + EW - 1) / EW), EW, 0, stream>>>(
      y3, x, g3, b3, sum3, sq3, out);
}
